// Enhanced_GAT_HSPA_76879914598912
// MI455X (gfx1250) — matrix operand order repaired and checked by reading, not compiled, not run on hardware
//
#include <hip/hip_runtime.h>
#include <hip/hip_bf16.h>

typedef _Float16 v16h __attribute__((ext_vector_type(16)));
typedef _Float16 h8   __attribute__((ext_vector_type(8)));
typedef float    v8f  __attribute__((ext_vector_type(8)));

#define WMMA_F16(a, b, c) \
  __builtin_amdgcn_wmma_f32_16x16x32_f16(false, (a), false, (b), (short)0, (c), false, false)

__device__ __forceinline__ v16h make_a_frag(const _Float16* ap) {
  // halves 0..7 = K[lh*8..+7], halves 8..15 = K[16+lh*8..+7]  (ISA 7.12.2, 16-bit A 16x32)
  h8 lo = *(const h8*)ap;
  h8 hi = *(const h8*)(ap + 16);
  return __builtin_shufflevector(lo, hi, 0, 1, 2, 3, 4, 5, 6, 7, 8, 9, 10, 11, 12, 13, 14, 15);
}

// Async global->LDS 16B copy per lane (ASYNCcnt tracked), SADDR addressing mode.
__device__ __forceinline__ void async_copy16(unsigned lds_off, unsigned goff, const void* base) {
  asm volatile("global_load_async_to_lds_b128 %0, %1, %2"
               :: "v"(lds_off), "v"(goff), "s"(base) : "memory");
}
__device__ __forceinline__ void async_wait() {
  asm volatile("s_wait_asynccnt 0x0" ::: "memory");
}

// ---------------------------------------------------------------- bn_prep
__global__ void bn_prep(const float* __restrict__ g, const float* __restrict__ be,
                        const float* __restrict__ m, const float* __restrict__ v,
                        const float* __restrict__ bias, float* __restrict__ scale,
                        float* __restrict__ shift, int C) {
  int i = blockIdx.x * blockDim.x + threadIdx.x;
  if (i < C) {
    float s = g[i] * rsqrtf(v[i] + 1e-5f);
    scale[i] = s;
    shift[i] = (bias[i] - m[i]) * s + be[i];
  }
}

// ---------------------------------------------------------------- embed: prelu(bn(W[128,256] @ X[256,N]))
// Whole-K panels resident in LDS (W: 64KB, B: 32KB), one barrier, then 8 K-steps x 4 N-subtiles
// of pure ds_load+wmma per wave. Output transposed f16: outT[b][n][c].
__global__ void embed_kernel(const float* __restrict__ X, const float* __restrict__ W,
                             const float* __restrict__ scale, const float* __restrict__ shift,
                             const float* __restrict__ alpha_p, _Float16* __restrict__ outT,
                             int N, int rowsT) {
  extern __shared__ _Float16 smem[];
  _Float16* sW = smem;                 // [128][256]
  _Float16* sB = smem + 128 * 256;     // [64][256]
  const int tid = threadIdx.x;
  const int wave = tid >> 5, lane = tid & 31;
  const int lr = lane & 15, lh = lane >> 4;
  const int n0 = blockIdx.x * 64;
  const int b = blockIdx.y;
  const float* Xb = X + (size_t)b * 256 * N;
  // stage W (f32 -> f16), vectorized 8-wide
  for (int i = tid; i < 128 * 256 / 8; i += 256) {
    int mm = i >> 5, c8 = (i & 31) * 8;
    const float* wp = &W[(size_t)mm * 256 + c8];
    h8 o;
    #pragma unroll
    for (int r = 0; r < 8; ++r) o[r] = (_Float16)wp[r];
    *(h8*)&sW[mm * 256 + c8] = o;
  }
  // stage B panel, transposed + converted; consecutive lanes read consecutive n (coalesced)
  for (int i = tid; i < 64 * 256; i += 256) {
    int nn = i & 63, kk = i >> 6;
    int gn = n0 + nn;
    sB[nn * 256 + kk] = (gn < N) ? (_Float16)Xb[(size_t)kk * N + gn] : (_Float16)0.0f;
  }
  __syncthreads();
  v8f acc[4] = {};
  for (int k0 = 0; k0 < 256; k0 += 32) {
    v16h a = make_a_frag(&sW[(wave * 16 + lr) * 256 + k0 + lh * 8]);
    #pragma unroll
    for (int ns = 0; ns < 4; ++ns) {
      v16h bf = make_a_frag(&sB[(ns * 16 + lr) * 256 + k0 + lh * 8]);
      acc[ns] = WMMA_F16(a, bf, acc[ns]);
    }
  }
  const float alpha = alpha_p[0];
  const int mbase = wave * 16 + lh * 8;
  #pragma unroll
  for (int ns = 0; ns < 4; ++ns) {
    int gn = n0 + ns * 16 + lr;
    if (gn < N) {
      h8 o;
      #pragma unroll
      for (int r = 0; r < 8; ++r) {
        float y = acc[ns][r] * scale[mbase + r] + shift[mbase + r];
        y = (y >= 0.0f) ? y : alpha * y;
        o[r] = (_Float16)y;
      }
      *(h8*)(outT + ((size_t)b * rowsT + gn) * 128 + mbase) = o;
    }
  }
}

// ---------------------------------------------------------------- sim = z_embT^T @ x_embT
// Both f16 panels copied with GLOBAL_LOAD_ASYNC_TO_LDS_B128; macro 64(M)x64(N),
// 8 waves = 4 M-strips x 2 N-halves, 2 accumulators per wave, K=128 fully resident.
__global__ void sim_kernel(const _Float16* __restrict__ zT, const _Float16* __restrict__ xT,
                           float* __restrict__ sim) {
  extern __shared__ _Float16 smem[];
  _Float16* sA = smem;               // [64][128]
  _Float16* sB = smem + 64 * 128;    // [64][128]
  const int tid = threadIdx.x;
  const int wave = tid >> 5, lane = tid & 31;
  const int lr = lane & 15, lh = lane >> 4;
  const int mw = wave & 3, nw = wave >> 2;
  const int n0 = blockIdx.x * 64;
  const int m0 = blockIdx.y * 64;
  const int b = blockIdx.z;
  const unsigned ldsA = (unsigned)(uintptr_t)sA;
  const unsigned ldsB = (unsigned)(uintptr_t)sB;
  const _Float16* gA = zT + ((size_t)b * 192 + m0) * 128;  // 64 rows x 256B, contiguous 16KB
  const _Float16* gB = xT + ((size_t)b * 640 + n0) * 128;  // contiguous 16KB
  #pragma unroll
  for (int u = 0; u < 4; ++u) {
    unsigned i = (unsigned)(tid + u * 256) * 16u;
    async_copy16(ldsA + i, i, gA);
  }
  #pragma unroll
  for (int u = 0; u < 4; ++u) {
    unsigned i = (unsigned)(tid + u * 256) * 16u;
    async_copy16(ldsB + i, i, gB);
  }
  async_wait();
  __syncthreads();
  v8f acc[2] = {};
  #pragma unroll
  for (int k0 = 0; k0 < 128; k0 += 32) {
    v16h a = make_a_frag(&sA[(mw * 16 + lr) * 128 + k0 + lh * 8]);
    #pragma unroll
    for (int ns = 0; ns < 2; ++ns) {
      v16h bf = make_a_frag(&sB[(nw * 32 + ns * 16 + lr) * 128 + k0 + lh * 8]);
      acc[ns] = WMMA_F16(a, bf, acc[ns]);
    }
  }
  const int mbase = m0 + mw * 16 + lh * 8;
  #pragma unroll
  for (int ns = 0; ns < 2; ++ns) {
    int n = n0 + nw * 32 + ns * 16 + lr;
    if (n < 625) {
      #pragma unroll
      for (int r = 0; r < 8; ++r) {
        int m = mbase + r;
        if (m < 169) sim[((size_t)b * 176 + m) * 640 + n] = acc[ns][r];
      }
    }
  }
}

// ---------------------------------------------------------------- sparsemax-topk per row, one wave32/row
__global__ void attn_kernel(const float* __restrict__ sim, _Float16* __restrict__ attnT) {
  const int wave = threadIdx.x >> 5, lane = threadIdx.x & 31;
  const int row = blockIdx.x * 8 + wave;
  if (row >= 256 * 169) return;
  const int b = row / 169, k = row % 169;
  const float* s = sim + ((size_t)b * 176 + k) * 640;
  float v[20], v0[20];
  float lmax = -3.0e38f;
  #pragma unroll
  for (int j = 0; j < 20; ++j) {
    int idx = lane * 20 + j;
    float x = (idx < 625) ? s[idx] : -3.0e38f;
    v[j] = x;
    lmax = fmaxf(lmax, x);
  }
  for (int off = 16; off >= 1; off >>= 1) lmax = fmaxf(lmax, __shfl_xor(lmax, off, 32));
  #pragma unroll
  for (int j = 0; j < 20; ++j) { v[j] -= lmax; v0[j] = v[j]; }
  float csum = 0.0f, csum_supp = 0.0f, suppf = 1.0f;
  for (int t = 1; t <= 64; ++t) {
    float lm = -3.0e38f;
    int li = 0;
    #pragma unroll
    for (int j = 0; j < 20; ++j)
      if (v[j] > lm) { lm = v[j]; li = j; }
    float gm = lm;
    for (int off = 16; off >= 1; off >>= 1) gm = fmaxf(gm, __shfl_xor(gm, off, 32));
    unsigned long long bal = __ballot(lm == gm);
    int first = (int)__ffsll(bal) - 1;
    if (lane == first) v[li] = -3.0e38f;  // remove exactly one copy (tie-safe)
    csum += gm;
    float ratio = csum / (gm + 1e-8f);
    if (t == 1) csum_supp = csum;
    if ((float)t <= ratio) { suppf = (float)t; csum_supp = csum; }
  }
  float tau = csum_supp / (suppf + 1e-8f);
  #pragma unroll
  for (int j = 0; j < 20; ++j) {
    int idx = lane * 20 + j;
    if (idx < 625) {
      float o = v0[j] - tau;
      o = (o > 0.0f) ? o : 0.0f;
      attnT[((size_t)b * 640 + idx) * 192 + k] = (_Float16)o;
    }
  }
}

// ---------------------------------------------------------------- agg = zf[256,169] @ attn[169,625]
// K padded 169->192 resident in LDS; zf staged synchronously (f32->f16 + zero-pad), attnT panel
// copied with async b128. Macro 128x64, 4 accumulators/wave, transposed f16 output.
__global__ void agg_kernel(const float* __restrict__ zf, const _Float16* __restrict__ attnT,
                           _Float16* __restrict__ aggT) {
  extern __shared__ _Float16 smem[];
  _Float16* sA = smem;               // [128][192]
  _Float16* sB = smem + 128 * 192;   // [64][192]
  const int tid = threadIdx.x;
  const int wave = tid >> 5, lane = tid & 31;
  const int lr = lane & 15, lh = lane >> 4;
  const int n0 = blockIdx.x * 64;
  const int m0 = blockIdx.y * 128;
  const int b = blockIdx.z;
  // async copy of attnT panel: 64 rows x 384B contiguous = 1536 x 16B chunks
  const unsigned ldsB = (unsigned)(uintptr_t)sB;
  const _Float16* gB = attnT + ((size_t)b * 640 + n0) * 192;
  #pragma unroll
  for (int u = 0; u < 6; ++u) {
    unsigned i = (unsigned)(tid + u * 256) * 16u;
    async_copy16(ldsB + i, i, gB);
  }
  // synchronous zf stage with K zero-padding (kills garbage in attnT pad columns)
  for (int i = tid; i < 128 * 192; i += 256) {
    int mm = i / 192, kk = i - mm * 192;
    sA[i] = (kk < 169) ? (_Float16)zf[((size_t)b * 256 + m0 + mm) * 169 + kk]
                       : (_Float16)0.0f;
  }
  async_wait();
  __syncthreads();
  v8f acc[4] = {};
  for (int k0 = 0; k0 < 192; k0 += 32) {
    v16h a = make_a_frag(&sA[(wave * 16 + lr) * 192 + k0 + lh * 8]);
    #pragma unroll
    for (int ns = 0; ns < 4; ++ns) {
      v16h bf = make_a_frag(&sB[(ns * 16 + lr) * 192 + k0 + lh * 8]);
      acc[ns] = WMMA_F16(a, bf, acc[ns]);
    }
  }
  const int mbase = m0 + wave * 16 + lh * 8;
  #pragma unroll
  for (int ns = 0; ns < 4; ++ns) {
    int gn = n0 + ns * 16 + lr;
    h8 o;
    #pragma unroll
    for (int r = 0; r < 8; ++r) o[r] = (_Float16)acc[ns][r];
    *(h8*)(aggT + ((size_t)b * 640 + gn) * 256 + mbase) = o;
  }
}

// ---------------------------------------------------------------- fused = prelu(bn(fu_w @ [agg;xf])),
// out = rs*fused + xf.  K=512 in 4 panels of 128; macro 128x64; 64 wmma per wave total.
__global__ void fuse_kernel(const float* __restrict__ xf, const _Float16* __restrict__ aggT,
                            const float* __restrict__ W, const float* __restrict__ scale,
                            const float* __restrict__ shift, const float* __restrict__ alpha_p,
                            const float* __restrict__ rs_p, float* __restrict__ out) {
  extern __shared__ _Float16 smem[];
  _Float16* sA = smem;               // [128][128]
  _Float16* sB = smem + 128 * 128;   // [64][128]
  const int tid = threadIdx.x;
  const int wave = tid >> 5, lane = tid & 31;
  const int lr = lane & 15, lh = lane >> 4;
  const int n0 = blockIdx.x * 64;
  const int m0 = blockIdx.y * 128;
  const int b = blockIdx.z;
  v8f acc[4] = {};
  for (int k0 = 0; k0 < 512; k0 += 128) {
    if (k0 + 128 < 512)  // speculative prefetch of next W panel (global_prefetch_b8)
      __builtin_prefetch(&W[(size_t)(m0 + wave * 16) * 512 + k0 + 128], 0, 0);
    for (int i = tid; i < 128 * 128; i += 256) {
      int mm = i >> 7, kk = i & 127;
      sA[i] = (_Float16)W[(size_t)(m0 + mm) * 512 + k0 + kk];
    }
    for (int i = tid; i < 64 * 128; i += 256) {
      int nn = i >> 7, kk = i & 127;
      int gk = k0 + kk, gn = n0 + nn;
      _Float16 val;
      if (gk < 256)
        val = aggT[((size_t)b * 640 + gn) * 256 + gk];
      else
        val = (gn < 625) ? (_Float16)xf[((size_t)b * 256 + (gk - 256)) * 625 + gn]
                         : (_Float16)0.0f;
      sB[nn * 128 + kk] = val;
    }
    __syncthreads();
    #pragma unroll
    for (int ks = 0; ks < 128; ks += 32) {
      v16h a = make_a_frag(&sA[(wave * 16 + lr) * 128 + ks + lh * 8]);
      #pragma unroll
      for (int ns = 0; ns < 4; ++ns) {
        v16h bf = make_a_frag(&sB[(ns * 16 + lr) * 128 + ks + lh * 8]);
        acc[ns] = WMMA_F16(a, bf, acc[ns]);
      }
    }
    __syncthreads();
  }
  const float alpha = alpha_p[0];
  const float rs = rs_p[0];
  const int mbase = m0 + wave * 16 + lh * 8;
  #pragma unroll
  for (int ns = 0; ns < 4; ++ns) {
    int gn = n0 + ns * 16 + lr;
    if (gn < 625) {
      #pragma unroll
      for (int r = 0; r < 8; ++r) {
        int ch = mbase + r;
        float y = acc[ns][r] * scale[ch] + shift[ch];
        y = (y >= 0.0f) ? y : alpha * y;
        size_t idx = ((size_t)b * 256 + ch) * 625 + gn;
        out[idx] = rs * y + xf[idx];
      }
    }
  }
}

// ---------------------------------------------------------------- launch
extern "C" void kernel_launch(void* const* d_in, const int* in_sizes, int n_in,
                              void* d_out, int out_size, void* d_ws, size_t ws_size,
                              hipStream_t stream) {
  const float* zf = (const float*)d_in[0];
  const float* xf = (const float*)d_in[1];
  const float* zt_w = (const float*)d_in[2];
  const float* zt_b = (const float*)d_in[3];
  const float* zt_g = (const float*)d_in[4];
  const float* zt_be = (const float*)d_in[5];
  const float* zt_m = (const float*)d_in[6];
  const float* zt_v = (const float*)d_in[7];
  const float* zt_a = (const float*)d_in[8];
  const float* xt_w = (const float*)d_in[9];
  const float* xt_b = (const float*)d_in[10];
  const float* xt_g = (const float*)d_in[11];
  const float* xt_be = (const float*)d_in[12];
  const float* xt_m = (const float*)d_in[13];
  const float* xt_v = (const float*)d_in[14];
  const float* xt_a = (const float*)d_in[15];
  const float* fu_w = (const float*)d_in[16];
  const float* fu_b = (const float*)d_in[17];
  const float* fu_g = (const float*)d_in[18];
  const float* fu_be = (const float*)d_in[19];
  const float* fu_m = (const float*)d_in[20];
  const float* fu_v = (const float*)d_in[21];
  const float* fu_a = (const float*)d_in[22];
  const float* res_scale = (const float*)d_in[23];
  float* out = (float*)d_out;

  char* ws = (char*)d_ws;
  float* zt_sc = (float*)ws;          // 128
  float* zt_sh = zt_sc + 128;
  float* xt_sc = zt_sh + 128;
  float* xt_sh = xt_sc + 128;
  float* fu_sc = xt_sh + 128;         // 256
  float* fu_sh = fu_sc + 256;         // 256  (< 4096 B total)
  size_t o_z = 4096;
  size_t o_x = o_z + (size_t)256 * 192 * 128 * 2;        // z_embT [B][192][128] f16
  size_t o_sim = o_x + (size_t)256 * 640 * 128 * 2;      // x_embT [B][640][128] f16
  size_t o_attn = o_sim + (size_t)256 * 176 * 640 * 4;   // sim    [B][176][640] f32
  _Float16* zT = (_Float16*)(ws + o_z);
  _Float16* xT = (_Float16*)(ws + o_x);
  float* simb = (float*)(ws + o_sim);
  _Float16* attnT = (_Float16*)(ws + o_attn);            // [B][640][192] f16
  _Float16* aggT = (_Float16*)(ws + o_sim);              // overlay on dead sim buffer

  bn_prep<<<1, 128, 0, stream>>>(zt_g, zt_be, zt_m, zt_v, zt_b, zt_sc, zt_sh, 128);
  bn_prep<<<1, 128, 0, stream>>>(xt_g, xt_be, xt_m, xt_v, xt_b, xt_sc, xt_sh, 128);
  bn_prep<<<1, 256, 0, stream>>>(fu_g, fu_be, fu_m, fu_v, fu_b, fu_sc, fu_sh, 256);

  const size_t shm_embed = (size_t)(128 * 256 + 64 * 256) * 2;  // 96 KB
  const size_t shm_sim   = (size_t)(64 * 128 + 64 * 128) * 2;   // 32 KB
  const size_t shm_agg   = (size_t)(128 * 192 + 64 * 192) * 2;  // 72 KB
  const size_t shm_fuse  = (size_t)(128 * 128 + 64 * 128) * 2;  // 48 KB

  embed_kernel<<<dim3(3, 256), 256, shm_embed, stream>>>(zf, zt_w, zt_sc, zt_sh, zt_a,
                                                         zT, 169, 192);
  embed_kernel<<<dim3(10, 256), 256, shm_embed, stream>>>(xf, xt_w, xt_sc, xt_sh, xt_a,
                                                          xT, 625, 640);
  sim_kernel<<<dim3(10, 3, 256), 256, shm_sim, stream>>>(zT, xT, simb);
  attn_kernel<<<(256 * 169 + 7) / 8, 256, 0, stream>>>(simb, attnT);
  agg_kernel<<<dim3(10, 2, 256), 256, shm_agg, stream>>>(zf, attnT, aggT);
  fuse_kernel<<<dim3(10, 2, 256), 256, shm_fuse, stream>>>(xf, aggT, fu_w, fu_sc, fu_sh,
                                                           fu_a, res_scale, out);
}
